// GraphAttentionLayer_14972255994171
// MI455X (gfx1250) — matrix operand order repaired and checked by reading, not compiled, not run on hardware
//
#include <hip/hip_runtime.h>

typedef __attribute__((ext_vector_type(16))) _Float16 v16h;
typedef __attribute__((ext_vector_type(8)))  float    v8f;

#define N_NODES 8192
#define FIN     128
#define FOUT    64
#define ALPHA   0.2f
#define LOG2E   1.4426950408889634f
#define TILES   32                 // (N_NODES/8)/32 per wave

// ---------------- Kernel 1: Wh = h @ W ----------------
__global__ __launch_bounds__(256) void k_gemm_wh(
    const float* __restrict__ h, const float* __restrict__ W,
    float* __restrict__ Whf, _Float16* __restrict__ Whc)
{
    int idx = blockIdx.x * 256 + threadIdx.x;   // N_NODES*FOUT threads
    int i = idx >> 6;
    int n = idx & 63;
    const float* hr = h + (size_t)i * FIN;
    float acc = 0.f;
#pragma unroll 8
    for (int k = 0; k < FIN; ++k)
        acc = fmaf(hr[k], W[k * FOUT + n], acc);
    Whf[(size_t)i * FOUT + n] = acc;
    Whc[(size_t)n * N_NODES + i] = (_Float16)acc;
}

// ---------------- Kernel 2: e1 = Wh@a1, e2 = Wh@a2 ----------------
__global__ __launch_bounds__(256) void k_e(
    const float* __restrict__ Whf, const float* __restrict__ a,
    float* __restrict__ e1, float* __restrict__ e2)
{
    int i = blockIdx.x * 256 + threadIdx.x;     // N_NODES threads
    const float* r = Whf + (size_t)i * FOUT;
    float s1 = 0.f, s2 = 0.f;
#pragma unroll
    for (int n = 0; n < FOUT; ++n) {
        float v = r[n];
        s1 = fmaf(v, a[n], s1);
        s2 = fmaf(v, a[FOUT + n], s2);
    }
    e1[i] = s1;
    e2[i] = s2;
}

// ---------------- Kernel 2b: global max of e2 (softmax shift bound) -------
__global__ __launch_bounds__(256) void k_e2max(const float* __restrict__ e2,
                                               float* __restrict__ e2max)
{
    __shared__ float red[256];
    float mx = -3.0e38f;
    for (int i = threadIdx.x; i < N_NODES; i += 256)
        mx = fmaxf(mx, e2[i]);
    red[threadIdx.x] = mx;
    __syncthreads();
    for (int s = 128; s > 0; s >>= 1) {
        if (threadIdx.x < s)
            red[threadIdx.x] = fmaxf(red[threadIdx.x], red[threadIdx.x + s]);
        __syncthreads();
    }
    if (threadIdx.x == 0) e2max[0] = red[0];
}

// ---------------- Kernel 3: fused masked softmax + attention@Wh -----------
// One block (8 waves) per 16-row tile; each wave streams 1/8 of the columns.
// m_i = leaky(e1[i]+max_j e2[j]) is a fixed per-row softmax shift (leaky is
// monotone), so partials over disjoint j are plain sums. Row sums l ride a
// 5th WMMA against an all-ones B fragment. adj/e2 use a 1-deep software
// pipeline + L2 prefetch; adj is read exactly once device-wide.
__global__ __launch_bounds__(256) void k_attn(
    const int* __restrict__ adj, const float* __restrict__ e1,
    const float* __restrict__ e2, const float* __restrict__ e2max,
    const _Float16* __restrict__ Whc, float* __restrict__ out)
{
    __shared__ float accbuf[8][32][40];   // [wave][lane][slot] 40 KB
                                          // slot 0..31: nb*8+r ; 32..39: l[r]
    const int lane = threadIdx.x & 31;
    const int wv   = threadIdx.x >> 5;
    const int i0   = blockIdx.x * 16;
    const int hh   = lane >> 4;                 // lane half (A-layout K group)
    const int M    = lane & 15;                 // row within block

    const float e1i  = e1[i0 + M];
    const float xm   = e1i + e2max[0];
    const float m    = fmaxf(xm, ALPHA * xm);   // fixed per-row softmax shift
    const float nmK  = -m * LOG2E;              // folded into exp2 argument
    const int* adjRow = adj + (size_t)(i0 + M) * N_NODES;

    v16h ones;
#pragma unroll
    for (int q = 0; q < 16; ++q) ones[q] = (_Float16)1.0f;

    v8f acc0 = {}, acc1 = {}, acc2 = {}, acc3 = {}, accL = {};

    const int j0 = wv * (N_NODES / 8);          // this wave's column range

    // ---- prologue: load tile 0 ----
    int cA = j0 + 16 * hh, cB = cA + 8;
    int4 ma0 = *(const int4*)(adjRow + cA);
    int4 ma1 = *(const int4*)(adjRow + cA + 4);
    int4 mb0 = *(const int4*)(adjRow + cB);
    int4 mb1 = *(const int4*)(adjRow + cB + 4);
    float4 ea0 = *(const float4*)(e2 + cA);
    float4 ea1 = *(const float4*)(e2 + cA + 4);
    float4 eb0 = *(const float4*)(e2 + cB);
    float4 eb1 = *(const float4*)(e2 + cB + 4);

    for (int it = 0; it < TILES; ++it) {        // uniform trip count
        const int jt  = j0 + it * 32;
        const int itn = (it + 1 < TILES) ? (it + 1) : it;   // clamp tail
        const int nA  = j0 + itn * 32 + 16 * hh;
        const int nB  = nA + 8;

        // prefetch the adjacency stream ~1KB ahead (speculative, OOB-safe)
        __builtin_prefetch(adjRow + jt + 8 * hh + 8 * 32, 0, 3);

        // issue next tile's adj/e2 loads (overlap with this tile's math)
        int4 xa0 = *(const int4*)(adjRow + nA);
        int4 xa1 = *(const int4*)(adjRow + nA + 4);
        int4 xb0 = *(const int4*)(adjRow + nB);
        int4 xb1 = *(const int4*)(adjRow + nB + 4);
        float4 fa0 = *(const float4*)(e2 + nA);
        float4 fa1 = *(const float4*)(e2 + nA + 4);
        float4 fb0 = *(const float4*)(e2 + nB);
        float4 fb1 = *(const float4*)(e2 + nB + 4);

        // B fragments: f16 column-major Wh, 32B contiguous per lane,
        // in contiguous k order 16*hh+s, same as pa (VGPR r: K=16*hh+2r, 16*hh+2r+1).
        const _Float16* bbase = Whc + jt + 16 * hh;
        v16h bf0 = *(const v16h*)(bbase + (size_t)(M +  0) * N_NODES);
        v16h bf1 = *(const v16h*)(bbase + (size_t)(M + 16) * N_NODES);
        v16h bf2 = *(const v16h*)(bbase + (size_t)(M + 32) * N_NODES);
        v16h bf3 = *(const v16h*)(bbase + (size_t)(M + 48) * N_NODES);

        int   msk[16] = {ma0.x,ma0.y,ma0.z,ma0.w, ma1.x,ma1.y,ma1.z,ma1.w,
                         mb0.x,mb0.y,mb0.z,mb0.w, mb1.x,mb1.y,mb1.z,mb1.w};
        float ev [16] = {ea0.x,ea0.y,ea0.z,ea0.w, ea1.x,ea1.y,ea1.z,ea1.w,
                         eb0.x,eb0.y,eb0.z,eb0.w, eb1.x,eb1.y,eb1.z,eb1.w};

        v16h pa;                                // A fragment (f16 16x32)
#pragma unroll
        for (int q = 0; q < 16; ++q) {
            float x  = e1i + ev[q];
            float lr = fmaxf(x, ALPHA * x);               // leaky_relu
            float p  = __builtin_exp2f(fmaf(lr, LOG2E, nmK));  // exp(lr-m)
            p = (msk[q] > 0) ? p : 0.f;
            pa[q] = (_Float16)p;
        }

        acc0 = __builtin_amdgcn_wmma_f32_16x16x32_f16(false, pa, false, bf0,  (short)0, acc0, false, false);
        acc1 = __builtin_amdgcn_wmma_f32_16x16x32_f16(false, pa, false, bf1,  (short)0, acc1, false, false);
        acc2 = __builtin_amdgcn_wmma_f32_16x16x32_f16(false, pa, false, bf2,  (short)0, acc2, false, false);
        acc3 = __builtin_amdgcn_wmma_f32_16x16x32_f16(false, pa, false, bf3,  (short)0, acc3, false, false);
        accL = __builtin_amdgcn_wmma_f32_16x16x32_f16(false, pa, false, ones, (short)0, accL, false, false);

        // rotate pipeline buffers
        ma0 = xa0; ma1 = xa1; mb0 = xb0; mb1 = xb1;
        ea0 = fa0; ea1 = fa1; eb0 = fb0; eb1 = fb1;
    }

    // publish this wave's partials (accL[r] = row sum, identical per column)
    float* dst = &accbuf[wv][lane][0];
    ((v8f*)dst)[0] = acc0;
    ((v8f*)dst)[1] = acc1;
    ((v8f*)dst)[2] = acc2;
    ((v8f*)dst)[3] = acc3;
    ((v8f*)dst)[4] = accL;
    __syncthreads();

    // Reduce 8 wave-partials: 256 threads x 4 outputs = 16x64 tile.
    const int t   = threadIdx.x;
    const int row = t >> 4;                     // 0..15
    const int c0  = (t & 15) * 4;               // 0,4,...,60
    const int hh2 = row >> 3;
    const int r   = row & 7;

    float lt = 0.f;
#pragma unroll
    for (int w = 0; w < 8; ++w) lt += accbuf[w][hh2 * 16][32 + r];
    const float inv = 1.0f / fmaxf(lt, 1e-38f);

#pragma unroll
    for (int k = 0; k < 4; ++k) {
        const int col  = c0 + k;
        const int nb   = col >> 4;
        const int ln   = hh2 * 16 + (col & 15); // lane that held (row,col)
        const int slot = nb * 8 + r;
        float s = 0.f;
#pragma unroll
        for (int w = 0; w < 8; ++w) s += accbuf[w][ln][slot];
        float x = s * inv;
        float o = (x > 0.f) ? x : (__expf(x) - 1.0f);   // elu
        out[(size_t)(i0 + row) * FOUT + col] = o;
    }
}

extern "C" void kernel_launch(void* const* d_in, const int* in_sizes, int n_in,
                              void* d_out, int out_size, void* d_ws, size_t ws_size,
                              hipStream_t stream) {
    const float* h   = (const float*)d_in[0];   // (8192,128)
    const int*   adj = (const int*)  d_in[1];   // (8192,8192)
    const float* W   = (const float*)d_in[2];   // (128,64)
    const float* a   = (const float*)d_in[3];   // (128,1)
    float* out = (float*)d_out;                 // (8192,64)

    char* ws = (char*)d_ws;
    float*    Whf   = (float*)ws;                        // 2 MB
    _Float16* Whc   = (_Float16*)(ws + 2097152);         // 1 MB
    float*    e1    = (float*)(ws + 3145728);            // 32 KB
    float*    e2    = (float*)(ws + 3145728 + 32768);    // 32 KB
    float*    e2max = (float*)(ws + 3145728 + 65536);    // 4 B

    k_gemm_wh<<<(N_NODES * FOUT) / 256, 256, 0, stream>>>(h, W, Whf, Whc);
    k_e<<<N_NODES / 256, 256, 0, stream>>>(Whf, a, e1, e2);
    k_e2max<<<1, 256, 0, stream>>>(e2, e2max);
    k_attn<<<N_NODES / 16, 256, 0, stream>>>(adj, e1, e2, e2max, Whc, out);
}
